// VAEnn_60129542229
// MI455X (gfx1250) — matrix operand order repaired and checked by reading, not compiled, not run on hardware
//
#include <hip/hip_runtime.h>

typedef __attribute__((ext_vector_type(16))) _Float16 v16h;
typedef __attribute__((ext_vector_type(8)))  float    v8f;

#define EPS_E 1e-3f
#define EPS_D 1e-5f

// ---------------- workspace layout (float offsets) ----------------
#define WS_XC     0u                        // [3][8192]
#define WS_X1     24576u                    // [64][8192]
#define WS_X2     548864u                   // [128][8192]
#define WS_CAPS   1597440u                  // [8][1024][16]
#define WS_CAPST  1728512u                  // [1024][128]
#define WS_ZMT    1859584u                  // [128][128]
#define WS_ZLT    1875968u                  // [128][128]
#define WS_ZMEAN  1892352u                  // [8][128]
#define WS_ZLVAR  1893376u                  // [8][128]
#define WS_PSUM   1894400u                  // [8][16384]
#define WS_PSQ    2025472u
#define WS_UMAX   2156544u
#define WS_UMIN   2287616u
#define WS_SCALE  2418688u                  // [16384]
#define WS_SHIFT  2435072u
#define WS_LATT   2451456u                  // [128][16384]
#define WS_D1     4548608u                  // [128][16384]
#define WS_D2     6645760u                  // [64][16384]
#define WS_D3     7694336u                  // [32][16384]
#define WS_MINA   8218624u                  // [8][1024]
#define WS_MINB   8226816u                  // [8][2048]
#define WS_WH     8243200u                  // 2097152 halves, fragment-swizzled
#define WS_XH     9291776u                  // 1048576 halves, [i][c]

#define PTS_SZ 49152
#define LAT_SZ 2097152

// ---------------- small transforms ----------------
__global__ void k_xc_transpose(const float* __restrict__ coords, float* __restrict__ xc) {
  int t = blockIdx.x * 256 + threadIdx.x;            // b*3072 + n*3 + k
  if (t >= 24576) return;
  int b = t / 3072, rem = t % 3072, n = rem / 3, k = rem % 3;
  xc[k * 8192 + b * 1024 + n] = coords[t];
}

// Pre-swizzle ppc weights into WMMA A-fragment order:
// AhS[o*128 + q*32 + hi*16 + e] = w[o*128 + k(e)],
// k(e) = q*32 + hi*16 + e.
// Each (o,q,hi) fragment is then 32 contiguous bytes -> clean b128 loads.
__global__ void k_w_to_half_swz(const float* __restrict__ w, _Float16* __restrict__ whs) {
  int t = blockIdx.x * 256 + threadIdx.x;
  if (t >= 2097152) return;
  int e  = t & 15;
  int hi = (t >> 4) & 1;
  int q  = (t >> 5) & 3;
  int o  = t >> 7;
  int v = e >> 1, h = e & 1;
  int k = q * 32 + hi * 16 + e;
  whs[t] = (_Float16)w[(size_t)o * 128 + k];
}

__global__ void k_x_to_half(const float* __restrict__ x2, _Float16* __restrict__ xh) {
  int t = blockIdx.x * 256 + threadIdx.x;            // t = i*128 + c
  if (t >= 1048576) return;
  int c = t & 127, i = t >> 7;
  xh[t] = (_Float16)x2[c * 8192 + i];
}

// ---------------- generic pointwise-conv + training BN + activation ----------------
// in: [K][I], out: [O][I]; grid.x == O; act: 0 = leaky-relu(0.3), 1 = relu
__global__ __launch_bounds__(256) void k_conv_bn_act(
    const float* __restrict__ in, const float* __restrict__ w,
    const float* __restrict__ bias, const float* __restrict__ g,
    const float* __restrict__ bb, float* __restrict__ out,
    int K, int I, float eps, int act) {
  int o = blockIdx.x, t = threadIdx.x;
  const float* wr = w + (size_t)o * K;
  float lsum = 0.f, lsq = 0.f;
  for (int i = t; i < I; i += 256) {
    float acc = bias[o];
    for (int c = 0; c < K; ++c) acc = fmaf(wr[c], in[(size_t)c * I + i], acc);
    out[(size_t)o * I + i] = acc;
    lsum += acc; lsq += acc * acc;
  }
  __shared__ float ssum[256], ssq[256];
  ssum[t] = lsum; ssq[t] = lsq; __syncthreads();
  for (int s = 128; s > 0; s >>= 1) {
    if (t < s) { ssum[t] += ssum[t + s]; ssq[t] += ssq[t + s]; }
    __syncthreads();
  }
  float inv = 1.f / (float)I;
  float mean = ssum[0] * inv;
  float var  = ssq[0] * inv - mean * mean;
  float sc = g[o] * rsqrtf(var + eps);
  float sh = bb[o] - mean * sc;
  for (int i = t; i < I; i += 256) {
    float y = sc * out[(size_t)o * I + i] + sh;
    out[(size_t)o * I + i] = act ? fmaxf(y, 0.f) : (y > 0.f ? y : 0.3f * y);
  }
}

// ---------------- PrimaryPointCaps GEMM via WMMA f16->f32 ----------------
// A: fragment-swizzled ppc weights f16. X: activations f16 [8192][128] (i-major).
// 4 waves/block; one wave per (b, 16-row o-tile); streams 64 i-tiles of batch b,
// reducing max/min/sum/sumsq per output row on the fly (never materializes u).
__global__ __launch_bounds__(128) void k_ppc_wmma(
    const _Float16* __restrict__ AhS, const _Float16* __restrict__ Xh,
    const float* __restrict__ bias,
    float* __restrict__ psum, float* __restrict__ psq,
    float* __restrict__ umax, float* __restrict__ umin) {
  int wid = blockIdx.x * 4 + (threadIdx.x >> 5);
  int b = wid >> 10;
  int ot = wid & 1023;
  int obase = ot << 4;
  int lane = threadIdx.x & 31;
  int mrow = lane & 15;
  int hi = lane >> 4;                       // 0 or 1 (half-wave)

  // Preload A fragments for all 4 K-chunks: one aligned 32B vector load each.
  v16h afrag[4];
  const _Float16* arow = AhS + (size_t)(obase + mrow) * 128 + (hi << 4);
#pragma unroll
  for (int q = 0; q < 4; ++q)
    afrag[q] = *(const v16h*)(arow + q * 32);

  float bias_lane[8];
#pragma unroll
  for (int r = 0; r < 8; ++r) bias_lane[r] = bias[obase + r + (hi << 3)];

  float rmax[8], rmin[8], rsum[8], rsq[8];
#pragma unroll
  for (int r = 0; r < 8; ++r) { rmax[r] = -3.4e38f; rmin[r] = 3.4e38f; rsum[r] = 0.f; rsq[r] = 0.f; }

#pragma unroll 2
  for (int it = 0; it < 64; ++it) {
    int icol = (b << 10) + (it << 4) + mrow;          // this lane's column
    const _Float16* xrow = Xh + (size_t)icol * 128 + (hi << 4);
    v16h bf[4];
#pragma unroll
    for (int q = 0; q < 4; ++q)                       // 8x global_load_b128 clause
      bf[q] = *(const v16h*)(xrow + q * 32);
    v8f acc = {};
#pragma unroll
    for (int q = 0; q < 4; ++q)
      acc = __builtin_amdgcn_wmma_f32_16x16x32_f16(false, afrag[q], false, bf[q],
                                                   (short)0, acc, false, false);
#pragma unroll
    for (int r = 0; r < 8; ++r) {                     // row m = r + hi*8, col = mrow
      float u = acc[r] + bias_lane[r];
      rmax[r] = fmaxf(rmax[r], u);
      rmin[r] = fminf(rmin[r], u);
      rsum[r] += u;
      rsq[r]  += u * u;
    }
  }
  // reduce across the 16 lanes that share each output row
#pragma unroll
  for (int off = 1; off < 16; off <<= 1) {
#pragma unroll
    for (int r = 0; r < 8; ++r) {
      rmax[r] = fmaxf(rmax[r], __shfl_xor(rmax[r], off, 32));
      rmin[r] = fminf(rmin[r], __shfl_xor(rmin[r], off, 32));
      rsum[r] += __shfl_xor(rsum[r], off, 32);
      rsq[r]  += __shfl_xor(rsq[r], off, 32);
    }
  }
  if (mrow == 0) {
#pragma unroll
    for (int r = 0; r < 8; ++r) {
      int o = obase + r + (hi << 3);
      size_t idx = (size_t)b * 16384 + o;
      umax[idx] = rmax[r]; umin[idx] = rmin[r];
      psum[idx] = rsum[r]; psq[idx]  = rsq[r];
    }
  }
}

__global__ void k_ppc_stats(const float* __restrict__ psum, const float* __restrict__ psq,
                            const float* __restrict__ g, const float* __restrict__ bb,
                            float* __restrict__ scale, float* __restrict__ shift) {
  int o = blockIdx.x * 256 + threadIdx.x;
  if (o >= 16384) return;
  float s = 0.f, q = 0.f;
  for (int b = 0; b < 8; ++b) { s += psum[b * 16384 + o]; q += psq[b * 16384 + o]; }
  float mean = s * (1.f / 8192.f);
  float var  = q * (1.f / 8192.f) - mean * mean;
  float sc = g[o] * rsqrtf(var + EPS_D);
  scale[o] = sc;
  shift[o] = bb[o] - mean * sc;
}

// max-after-BN (affine) + squash over the 16 capsule channels
__global__ void k_squash(const float* __restrict__ scale, const float* __restrict__ shift,
                         const float* __restrict__ umax, const float* __restrict__ umin,
                         float* __restrict__ caps) {
  int t = blockIdx.x * 256 + threadIdx.x;            // b*1024 + cap
  if (t >= 8192) return;
  int b = t >> 10, cap = t & 1023;
  float v[16]; float sq = 0.f;
#pragma unroll
  for (int k = 0; k < 16; ++k) {
    int o = k * 1024 + cap;
    float sc = scale[o];
    float val = (sc > 0.f ? sc * umax[(size_t)b * 16384 + o]
                          : sc * umin[(size_t)b * 16384 + o]) + shift[o];
    v[k] = val; sq += val * val;
  }
  float f = sq / ((1.f + sq) * sqrtf(sq));
#pragma unroll
  for (int k = 0; k < 16; ++k) caps[(size_t)b * 16384 + cap * 16 + k] = v[k] * f;
}

__global__ void k_caps_transpose(const float* __restrict__ caps, float* __restrict__ capsT) {
  int t = blockIdx.x * 256 + threadIdx.x;            // b*16384 + c*16 + k
  if (t >= 131072) return;
  int b = t >> 14, rem = t & 16383, c = rem >> 4, k = rem & 15;
  capsT[c * 128 + b * 16 + k] = caps[t];
}

// FC [8,2048]->[8,128] + training BN over batch. zT layout: [o][b*16+i], f = o*16+i.
__global__ __launch_bounds__(256) void k_efc_bn(
    const float* __restrict__ zT, const float* __restrict__ W,
    const float* __restrict__ bias, const float* __restrict__ g,
    const float* __restrict__ bb, float* __restrict__ zout) {
  int l = blockIdx.x, t = threadIdx.x;
  int b = t >> 5, lane = t & 31;
  float part = 0.f;
  for (int f = lane; f < 2048; f += 32) {
    float zv = zT[(f >> 4) * 128 + b * 16 + (f & 15)];
    part = fmaf(zv, W[(size_t)l * 2048 + f], part);
  }
  for (int off = 16; off > 0; off >>= 1) part += __shfl_xor(part, off, 32);
  __shared__ float tb[8];
  if (lane == 0) tb[b] = part + bias[l];
  __syncthreads();
  if (t == 0) {
    float s = 0.f, q = 0.f;
    for (int j = 0; j < 8; ++j) { s += tb[j]; q += tb[j] * tb[j]; }
    float mean = s * 0.125f, var = q * 0.125f - mean * mean;
    float sc = g[l] * rsqrtf(var + EPS_D), sh = bb[l] - mean * sc;
    for (int j = 0; j < 8; ++j) zout[j * 128 + l] = sc * tb[j] + sh;
  }
}

__global__ void k_latent(const float* __restrict__ zm, const float* __restrict__ zlv,
                         const float* __restrict__ eps_in, float* __restrict__ lat) {
  size_t t = (size_t)blockIdx.x * 256 + threadIdx.x; // b*262144 + l*2048 + m
  if (t >= (size_t)LAT_SZ) return;
  size_t bl = t >> 11;
  lat[t] = zm[bl] + expf(zlv[bl]) * eps_in[t];
}

__global__ void k_lat_transpose(const float* __restrict__ lat, float* __restrict__ latT) {
  size_t t = (size_t)blockIdx.x * 256 + threadIdx.x;
  if (t >= (size_t)LAT_SZ) return;
  size_t b = t >> 18, rem = t & 262143, c = rem >> 11, m = rem & 2047;
  latT[c * 16384 + b * 2048 + m] = lat[t];
}

__global__ void k_decfc_tanh(const float* __restrict__ d3, const float* __restrict__ w,
                             const float* __restrict__ bias, float* __restrict__ pts) {
  int i = blockIdx.x * 256 + threadIdx.x;            // b*2048 + m
  if (i >= 16384) return;
  int b = i >> 11, m = i & 2047;
  float x[32];
#pragma unroll
  for (int c = 0; c < 32; ++c) x[c] = d3[c * 16384 + i];
#pragma unroll
  for (int j = 0; j < 3; ++j) {
    float acc = bias[j];
#pragma unroll
    for (int c = 0; c < 32; ++c) acc = fmaf(w[j * 32 + c], x[c], acc);
    pts[(size_t)b * 6144 + m * 3 + j] = tanhf(acc);
  }
}

__global__ void k_chamferA(const float* __restrict__ coords, const float* __restrict__ pts,
                           float* __restrict__ minA) {
  int t = blockIdx.x * 256 + threadIdx.x;            // b*1024 + n
  if (t >= 8192) return;
  int b = t >> 10, n = t & 1023;
  float x0 = coords[b * 3072 + n * 3 + 0];
  float x1 = coords[b * 3072 + n * 3 + 1];
  float x2 = coords[b * 3072 + n * 3 + 2];
  const float* z = pts + (size_t)b * 6144;
  float mn = 3.4e38f;
  for (int m0 = 0; m0 < 2048; m0 += 64) {
    __builtin_prefetch(z + (m0 + 64) * 3, 0, 1);     // global_prefetch_b8
    for (int m = m0; m < m0 + 64; ++m) {
      float d0 = x0 - z[m * 3], d1 = x1 - z[m * 3 + 1], d2 = x2 - z[m * 3 + 2];
      mn = fminf(mn, fmaf(d0, d0, fmaf(d1, d1, d2 * d2)));
    }
  }
  minA[t] = mn;
}

__global__ void k_chamferB(const float* __restrict__ coords, const float* __restrict__ pts,
                           float* __restrict__ minB) {
  int t = blockIdx.x * 256 + threadIdx.x;            // b*2048 + m
  if (t >= 16384) return;
  int b = t >> 11, m = t & 2047;
  float z0 = pts[(size_t)b * 6144 + m * 3 + 0];
  float z1 = pts[(size_t)b * 6144 + m * 3 + 1];
  float z2 = pts[(size_t)b * 6144 + m * 3 + 2];
  const float* x = coords + (size_t)b * 3072;
  float mn = 3.4e38f;
  for (int n0 = 0; n0 < 1024; n0 += 64) {
    __builtin_prefetch(x + (n0 + 64) * 3, 0, 1);     // global_prefetch_b8
    for (int n = n0; n < n0 + 64; ++n) {
      float d0 = z0 - x[n * 3], d1 = z1 - x[n * 3 + 1], d2 = z2 - x[n * 3 + 2];
      mn = fminf(mn, fmaf(d0, d0, fmaf(d1, d1, d2 * d2)));
    }
  }
  minB[t] = mn;
}

__global__ __launch_bounds__(256) void k_loss(const float* __restrict__ minA,
                                              const float* __restrict__ minB,
                                              const float* __restrict__ zm,
                                              const float* __restrict__ zlv,
                                              float* __restrict__ loss) {
  int t = threadIdx.x;
  float acc = 0.f;
  for (int i = t; i < 8192;  i += 256) acc += minA[i] * (1.f / (1024.f * 8.f));
  for (int i = t; i < 16384; i += 256) acc += minB[i] * (1.f / (2048.f * 8.f));
  for (int i = t; i < 1024;  i += 256) {
    float m = zm[i], v = zlv[i];
    acc += (-0.0005f / (128.f * 8.f)) * (1.f + v - m * m - expf(v));
  }
  __shared__ float sh[256];
  sh[t] = acc; __syncthreads();
  for (int s = 128; s > 0; s >>= 1) { if (t < s) sh[t] += sh[t + s]; __syncthreads(); }
  if (t == 0) *loss = sh[0];
}

// ---------------- launcher ----------------
extern "C" void kernel_launch(void* const* d_in, const int* in_sizes, int n_in,
                              void* d_out, int out_size, void* d_ws, size_t ws_size,
                              hipStream_t stream) {
  const float* coords = (const float*)d_in[0];
  const float* eps_in = (const float*)d_in[1];
  const float* P[44];
  for (int i = 2; i < 44 && i < n_in; ++i) P[i] = (const float*)d_in[i];

  float* ws = (float*)d_ws;
  float* out = (float*)d_out;
  float* lat_out = out + PTS_SZ;
  float* loss_out = out + PTS_SZ + LAT_SZ;

  _Float16* whs = (_Float16*)(ws + WS_WH);
  _Float16* xh  = (_Float16*)(ws + WS_XH);

  // encoder
  k_xc_transpose<<<96, 256, 0, stream>>>(coords, ws + WS_XC);
  k_conv_bn_act<<<64, 256, 0, stream>>>(ws + WS_XC, P[2], P[3], P[4], P[5],
                                        ws + WS_X1, 3, 8192, EPS_E, 0);
  k_conv_bn_act<<<128, 256, 0, stream>>>(ws + WS_X1, P[6], P[7], P[8], P[9],
                                         ws + WS_X2, 64, 8192, EPS_E, 0);

  // PrimaryPointCaps (WMMA f16)
  k_w_to_half_swz<<<8192, 256, 0, stream>>>(P[10], whs);
  k_x_to_half<<<4096, 256, 0, stream>>>(ws + WS_X2, xh);
  k_ppc_wmma<<<2048, 128, 0, stream>>>(whs, xh, P[11],
                                       ws + WS_PSUM, ws + WS_PSQ,
                                       ws + WS_UMAX, ws + WS_UMIN);
  k_ppc_stats<<<64, 256, 0, stream>>>(ws + WS_PSUM, ws + WS_PSQ, P[12], P[13],
                                      ws + WS_SCALE, ws + WS_SHIFT);
  k_squash<<<32, 256, 0, stream>>>(ws + WS_SCALE, ws + WS_SHIFT,
                                   ws + WS_UMAX, ws + WS_UMIN, ws + WS_CAPS);
  k_caps_transpose<<<512, 256, 0, stream>>>(ws + WS_CAPS, ws + WS_CAPST);

  // mean / logvar heads
  k_conv_bn_act<<<128, 256, 0, stream>>>(ws + WS_CAPST, P[14], P[15], P[16], P[17],
                                         ws + WS_ZMT, 1024, 128, EPS_E, 0);
  k_conv_bn_act<<<128, 256, 0, stream>>>(ws + WS_CAPST, P[18], P[19], P[20], P[21],
                                         ws + WS_ZLT, 1024, 128, EPS_E, 0);
  k_efc_bn<<<128, 256, 0, stream>>>(ws + WS_ZMT, P[22], P[23], P[24], P[25], ws + WS_ZMEAN);
  k_efc_bn<<<128, 256, 0, stream>>>(ws + WS_ZLT, P[26], P[27], P[28], P[29], ws + WS_ZLVAR);

  // latent (also an output)
  k_latent<<<8192, 256, 0, stream>>>(ws + WS_ZMEAN, ws + WS_ZLVAR, eps_in, lat_out);
  k_lat_transpose<<<8192, 256, 0, stream>>>(lat_out, ws + WS_LATT);

  // decoder
  k_conv_bn_act<<<128, 256, 0, stream>>>(ws + WS_LATT, P[30], P[31], P[32], P[33],
                                         ws + WS_D1, 128, 16384, EPS_D, 1);
  k_conv_bn_act<<<64, 256, 0, stream>>>(ws + WS_D1, P[34], P[35], P[36], P[37],
                                        ws + WS_D2, 128, 16384, EPS_D, 1);
  k_conv_bn_act<<<32, 256, 0, stream>>>(ws + WS_D2, P[38], P[39], P[40], P[41],
                                        ws + WS_D3, 64, 16384, EPS_D, 1);
  k_decfc_tanh<<<64, 256, 0, stream>>>(ws + WS_D3, P[42], P[43], out);

  // chamfer + loss
  k_chamferA<<<32, 256, 0, stream>>>(coords, out, ws + WS_MINA);
  k_chamferB<<<64, 256, 0, stream>>>(coords, out, ws + WS_MINB);
  k_loss<<<1, 256, 0, stream>>>(ws + WS_MINA, ws + WS_MINB,
                                ws + WS_ZMEAN, ws + WS_ZLVAR, loss_out);
}
